// GCN_10703058501716
// MI455X (gfx1250) — compile-verified, matrix operand order checked by reading, not run on hardware
//
#include <hip/hip_runtime.h>

// ---------------------------------------------------------------------------
// GCN forward on MI455X (gfx1250, wave32):
//   H1 = X @ W1            (bf16 WMMA, f32 accum, fragment-swizzled operands)
//   agg1 = scatter_add(ew * H1[src]) ; relu(agg1 + b1)
//   H2 = agg1 @ W2         (bf16 WMMA, N padded 40->48)
//   agg2 = scatter_add(ew * H2[src])
//   out = log_softmax(agg2 + b2)
// ---------------------------------------------------------------------------

typedef __attribute__((ext_vector_type(16))) __bf16 v16bf;
typedef __attribute__((ext_vector_type(8)))  float  v8f;

#define NNODES 100000
#define NFEAT  512
#define NHID   128
#define NCLASS 40
#define NCLASS_PAD 48
#define KS1 (NFEAT / 32)    // 16 K-steps, layer 1
#define KS2 (NHID  / 32)    // 4  K-steps, layer 2

__device__ __forceinline__ unsigned pack_bf16(float x, float y) {
    union { unsigned u; __bf16 h[2]; } p;
    p.h[0] = (__bf16)x;     // lowers to v_cvt_pk_bf16_f32
    p.h[1] = (__bf16)y;
    return p.u;
}

// ---------------------------------------------------------------- zero fill
__global__ void gcn_zero_f32x4(float4* __restrict__ p, long n4) {
    long i = (long)blockIdx.x * blockDim.x + threadIdx.x;
    if (i < n4) p[i] = make_float4(0.f, 0.f, 0.f, 0.f);
}

// -------------------------------------------- W1 -> bf16 fragment swizzle
// Dest layout: [ntile(8)][kstep(16)][lane(32)][8 dwords] = 128 KB (L2 resident)
// Fragment dword p of lane holds K = 32*ks + 2p + 16*(lane>=16), N = nt*16+(lane&15).
__global__ void gcn_pack_w1(const float* __restrict__ W, unsigned* __restrict__ Wf) {
    int t = blockIdx.x * blockDim.x + threadIdx.x;   // (512/2)*128 = 32768
    if (t >= (NFEAT / 2) * NHID) return;
    int n  = t & (NHID - 1);
    int kp = t >> 7;                 // K pair index 0..255
    int k  = kp * 2;
    float x = W[(long)k * NHID + n];
    float y = W[(long)(k + 1) * NHID + n];
    int ks = kp >> 4, half = (kp >> 3) & 1, p = kp & 7;
    int lane = (n & 15) + 16 * half;
    int nt   = n >> 4;
    Wf[(((long)nt * KS1 + ks) * 32 + lane) * 8 + p] = pack_bf16(x, y);
}

// -------------------------------------------- W2 (pad 40->48) -> bf16 swizzle
// Dest layout: [ntile(3)][kstep(4)][lane(32)][8 dwords] = 12 KB
__global__ void gcn_pack_w2(const float* __restrict__ W, unsigned* __restrict__ Wf) {
    int t = blockIdx.x * blockDim.x + threadIdx.x;   // (128/2)*48 = 3072
    if (t >= (NHID / 2) * NCLASS_PAD) return;
    int n  = t % NCLASS_PAD;
    int kp = t / NCLASS_PAD;         // 0..63
    int k  = kp * 2;
    float x = (n < NCLASS) ? W[(long)k * NCLASS + n] : 0.0f;
    float y = (n < NCLASS) ? W[(long)(k + 1) * NCLASS + n] : 0.0f;
    int ks = kp >> 4, half = (kp >> 3) & 1, p = kp & 7;
    int lane = (n & 15) + 16 * half;
    int nt   = n >> 4;
    Wf[(((long)nt * KS2 + ks) * 32 + lane) * 8 + p] = pack_bf16(x, y);
}

// ------------------------------------------------- GEMM1: [16,512]x[512,128]
// Block = 256 threads = 8 waves; wave w owns N-tile w. A panel staged into LDS
// pre-converted & fragment-swizzled; K loop fully unrolled -> 16 WMMAs with
// 2x ds_load_b128 + 2x global_load_b128 each.
__global__ __launch_bounds__(256) void gcn_gemm1_wmma(
        const float* __restrict__ X,    // [NNODES, NFEAT] f32
        const __bf16* __restrict__ Wf,  // swizzled W1 fragments
        float* __restrict__ H)          // [NNODES, NHID]
{
    __shared__ __align__(32) unsigned ldsA[KS1 * 32 * 8];   // 16 KB
    const int mt   = blockIdx.x;
    const int lane = threadIdx.x & 31;
    const int wv   = threadIdx.x >> 5;    // 0..7 -> N tile

    // warm cache lines holding this wave's first B fragment
    __builtin_prefetch(Wf + (((long)wv * KS1) * 32 + lane) * 16, 0, 0);

    // stage + convert + swizzle the 16x512 A panel (coalesced float2 reads)
    const float2* Xp2 = (const float2*)(X + (long)mt * 16 * NFEAT);
    for (int g = threadIdx.x; g < 16 * (NFEAT / 2); g += 256) {
        int m  = g >> 8;                  // row 0..15  (NFEAT/2 == 256)
        int kp = g & 255;                 // K pair
        float2 xy = Xp2[m * (NFEAT / 2) + kp];
        int ks = kp >> 4, half = (kp >> 3) & 1, p = kp & 7;
        int dl = m + 16 * half;
        ldsA[(ks * 32 + dl) * 8 + p] = pack_bf16(xy.x, xy.y);
    }
    __syncthreads();

    const int half = lane >> 4;
    const int l15  = lane & 15;
    const int ncol = wv * 16 + l15;

    v8f c = {0.f, 0.f, 0.f, 0.f, 0.f, 0.f, 0.f, 0.f};
    #pragma unroll
    for (int ks = 0; ks < KS1; ++ks) {
        v16bf a = *(const v16bf*)&ldsA[(ks * 32 + lane) * 8];
        v16bf b = *(const v16bf*)(Wf + (((long)wv * KS1 + ks) * 32 + lane) * 16);
        c = __builtin_amdgcn_wmma_f32_16x16x32_bf16(
                false, a, false, b, (short)0, c, false, false);
    }

    #pragma unroll
    for (int r = 0; r < 8; ++r)
        H[((long)mt * 16 + r + 8 * half) * NHID + ncol] = c[r];
}

// ------------------------------------------------- GEMM2: [16,128]x[128,48]
// Block = 96 threads = 3 waves (N = 48 = 3*16).
__global__ __launch_bounds__(96) void gcn_gemm2_wmma(
        const float* __restrict__ X,    // [NNODES, NHID] f32 (post relu)
        const __bf16* __restrict__ Wf,  // swizzled W2 fragments
        float* __restrict__ H)          // [NNODES, NCLASS_PAD]
{
    __shared__ __align__(32) unsigned ldsA[KS2 * 32 * 8];   // 4 KB
    const int mt   = blockIdx.x;
    const int lane = threadIdx.x & 31;
    const int wv   = threadIdx.x >> 5;    // 0..2

    __builtin_prefetch(Wf + (((long)wv * KS2) * 32 + lane) * 16, 0, 0);

    const float2* Xp2 = (const float2*)(X + (long)mt * 16 * NHID);
    for (int g = threadIdx.x; g < 16 * (NHID / 2); g += 96) {
        int m  = g >> 6;                  // NHID/2 == 64
        int kp = g & 63;
        float2 xy = Xp2[m * (NHID / 2) + kp];
        int ks = kp >> 4, half = (kp >> 3) & 1, p = kp & 7;
        int dl = m + 16 * half;
        ldsA[(ks * 32 + dl) * 8 + p] = pack_bf16(xy.x, xy.y);
    }
    __syncthreads();

    const int half = lane >> 4;
    const int l15  = lane & 15;
    const int ncol = wv * 16 + l15;

    v8f c = {0.f, 0.f, 0.f, 0.f, 0.f, 0.f, 0.f, 0.f};
    #pragma unroll
    for (int ks = 0; ks < KS2; ++ks) {
        v16bf a = *(const v16bf*)&ldsA[(ks * 32 + lane) * 8];
        v16bf b = *(const v16bf*)(Wf + (((long)wv * KS2 + ks) * 32 + lane) * 16);
        c = __builtin_amdgcn_wmma_f32_16x16x32_bf16(
                false, a, false, b, (short)0, c, false, false);
    }

    #pragma unroll
    for (int r = 0; r < 8; ++r)
        H[((long)mt * 16 + r + 8 * half) * NCLASS_PAD + ncol] = c[r];
}

// ------------------------------------------------- SpMM layer 1 (128 cols)
// 8 edges per 256-thread block; each lane gathers a float4 and fires 4
// hardware f32 atomics (global_atomic_add_f32).
__global__ __launch_bounds__(256) void gcn_spmm_scatter128(
        const float* __restrict__ H, const int* __restrict__ src,
        const int* __restrict__ dst, const float* __restrict__ ew,
        float* __restrict__ AGG, int E)
{
    int e = blockIdx.x * 8 + (threadIdx.x >> 5);
    if (e >= E) return;
    int lane = threadIdx.x & 31;
    int s = src[e], d = dst[e];
    float w = ew[e];
    float4 h = ((const float4*)(H + (long)s * NHID))[lane];
    float* Ar = AGG + (long)d * NHID + lane * 4;
    unsafeAtomicAdd(Ar + 0, w * h.x);
    unsafeAtomicAdd(Ar + 1, w * h.y);
    unsafeAtomicAdd(Ar + 2, w * h.z);
    unsafeAtomicAdd(Ar + 3, w * h.w);
}

// ------------------------------------------------- SpMM layer 2 (40 cols)
__global__ void gcn_spmm_scatter40(
        const float* __restrict__ H, const int* __restrict__ src,
        const int* __restrict__ dst, const float* __restrict__ ew,
        float* __restrict__ AGG)
{
    const int e = blockIdx.x;
    const int c = threadIdx.x;
    if (c >= NCLASS) return;
    const int s = src[e], d = dst[e];
    const float v = ew[e] * H[(long)s * NCLASS_PAD + c];
    unsafeAtomicAdd(&AGG[(long)d * NCLASS + c], v);
}

// ------------------------------------------------- bias + relu (in place)
__global__ void gcn_bias_relu4(float4* __restrict__ A, const float4* __restrict__ b,
                               long n4)
{
    long i = (long)blockIdx.x * blockDim.x + threadIdx.x;
    if (i < n4) {
        float4 v = A[i];
        float4 bb = b[(int)(i & (NHID / 4 - 1))];   // row length 32 float4s
        v.x = fmaxf(v.x + bb.x, 0.f);
        v.y = fmaxf(v.y + bb.y, 0.f);
        v.z = fmaxf(v.z + bb.z, 0.f);
        v.w = fmaxf(v.w + bb.w, 0.f);
        A[i] = v;
    }
}

// ------------------------------------------------- bias + log_softmax (row)
__global__ void gcn_bias_logsoftmax(const float* __restrict__ A,
                                    const float* __restrict__ b,
                                    float* __restrict__ out)
{
    int r = blockIdx.x * blockDim.x + threadIdx.x;
    if (r >= NNODES) return;
    float v[NCLASS];
    float mx = -3.0e38f;
    #pragma unroll
    for (int c = 0; c < NCLASS; ++c) {
        v[c] = A[(long)r * NCLASS + c] + b[c];
        mx = fmaxf(mx, v[c]);
    }
    float s = 0.0f;
    #pragma unroll
    for (int c = 0; c < NCLASS; ++c) s += __expf(v[c] - mx);
    float ls = __logf(s) + mx;
    #pragma unroll
    for (int c = 0; c < NCLASS; ++c) out[(long)r * NCLASS + c] = v[c] - ls;
}

// ---------------------------------------------------------------------------
extern "C" void kernel_launch(void* const* d_in, const int* in_sizes, int n_in,
                              void* d_out, int out_size, void* d_ws, size_t ws_size,
                              hipStream_t stream)
{
    const float* features = (const float*)d_in[0];
    const int*   esrc     = (const int*)  d_in[1];
    const int*   edst     = (const int*)  d_in[2];
    const float* ew       = (const float*)d_in[3];
    const float* W1       = (const float*)d_in[4];
    const float* b1       = (const float*)d_in[5];
    const float* W2       = (const float*)d_in[6];
    const float* b2       = (const float*)d_in[7];
    float*       out      = (float*)d_out;
    const int    E        = in_sizes[1];

    // Workspace layout (aliased; ~102.6 MB):
    //   [0 .. 51.2MB)       H1 f32 (100000x128) -> later H2 f32 (100000x48)
    //   [51.2 .. 102.4MB)   agg1 f32 (100000x128) -> later agg2 (100000x40)
    //   [102.4MB .. +128KB) W1 bf16 fragments
    //   [.. +12KB)          W2 bf16 fragments
    char* ws = (char*)d_ws;
    const size_t offAgg = (size_t)NNODES * NHID * sizeof(float);
    const size_t offW1  = offAgg * 2;
    const size_t offW2  = offW1 + (size_t)(NFEAT * NHID) * sizeof(__bf16);

    float*    H1    = (float*)ws;
    float*    AGG1  = (float*)(ws + offAgg);
    unsigned* Wf1u  = (unsigned*)(ws + offW1);
    unsigned* Wf2u  = (unsigned*)(ws + offW2);
    float*    H2    = H1;
    float*    AGG2  = AGG1;

    const long n1 = (long)NNODES * NHID;
    const long n2 = (long)NNODES * NCLASS;

    // Weight fragment packing (tiny, once per launch, L2 resident afterwards)
    gcn_pack_w1<<<(32768 + 255) / 256, 256, 0, stream>>>(W1, Wf1u);
    gcn_pack_w2<<<(3072 + 255) / 256, 256, 0, stream>>>(W2, Wf2u);

    // Layer 1
    gcn_zero_f32x4<<<(int)((n1 / 4 + 255) / 256), 256, 0, stream>>>(
        (float4*)AGG1, n1 / 4);
    gcn_gemm1_wmma<<<NNODES / 16, 256, 0, stream>>>(
        features, (const __bf16*)Wf1u, H1);
    gcn_spmm_scatter128<<<(E + 7) / 8, 256, 0, stream>>>(
        H1, esrc, edst, ew, AGG1, E);
    gcn_bias_relu4<<<(int)((n1 / 4 + 255) / 256), 256, 0, stream>>>(
        (float4*)AGG1, (const float4*)b1, n1 / 4);

    // Layer 2 (H2 aliases H1 — H1 dead after spmm1; agg2 aliases agg1 —
    // zeroed only after gemm2 consumed agg1, stream-ordered).
    gcn_gemm2_wmma<<<NNODES / 16, 96, 0, stream>>>(
        AGG1, (const __bf16*)Wf2u, H2);
    gcn_zero_f32x4<<<(int)((n2 / 4 + 255) / 256), 256, 0, stream>>>(
        (float4*)AGG2, n2 / 4);
    gcn_spmm_scatter40<<<E, 64, 0, stream>>>(H2, esrc, edst, ew, AGG2);
    gcn_bias_logsoftmax<<<(NNODES + 63) / 64, 64, 0, stream>>>(AGG2, b2, out);
}
